// GQA_3951369912410
// MI455X (gfx1250) — matrix operand order repaired and checked by reading, not compiled, not run on hardware
//
#include <hip/hip_runtime.h>
#include <hip/hip_bf16.h>
#include <math.h>
#include <stdint.h>

// ---------------------------------------------------------------------------
// GQA collapsed form (group axis g is summed in both einsums):
//   Ksum = x @ W1k  (W1k[e_in][e] = sum_g W1[e_in][g*128+e])        [B*S,128]
//   Vsum = x @ W1v                                                   [B*S,128]
//   Q    = x @ W2                                                    [B*S,2048]
//   S    = Q_head @ Ksum^T  (+mask) -> softmax -> @ Vsum -> O        [B*S,2048]
//   out  = O @ W3 (fp32)
// ~142 GFLOP vs ~150MB essential traffic @ 23.3TB/s => compute bound =>
// bf16 WMMA everywhere; GEMM LDS staging via Tensor Data Mover (TDM).
// ---------------------------------------------------------------------------

typedef __attribute__((ext_vector_type(16))) __bf16        v16bf;
typedef __attribute__((ext_vector_type(8)))  float         v8f;
typedef __attribute__((ext_vector_type(4)))  unsigned int  v4u;
typedef __attribute__((ext_vector_type(8)))  int           v8i;
typedef __attribute__((ext_vector_type(4)))  int           v4i;

#define B_   2
#define S_   2048
#define E_   2048
#define H_   16
#define G_   4
#define HD_  128
#define MT_  (B_ * S_)   // 4096 total rows

#if __has_builtin(__builtin_amdgcn_tensor_load_to_lds) && \
    __has_builtin(__builtin_amdgcn_s_wait_tensorcnt)
#define HAVE_TDM 1
#else
#define HAVE_TDM 0
#endif

union FragBF {          // one WMMA 16-bit operand = 8 VGPRs = 2x 16B chunks
    v16bf v;
    uint4 q[2];
};

static __device__ inline v8f vzero8() {
    v8f z = {0.f, 0.f, 0.f, 0.f, 0.f, 0.f, 0.f, 0.f};
    return z;
}

#if HAVE_TDM
// Issue one TDM 2D tile load: global (row-major, stride0 4B-units) -> LDS.
// Tile = tile_d0 (contiguous, 4B units) x tile_d1 rows. LDS rows get
// 16B padding after each 64B of data (pad_interval=16 DW, pad_amount=4 DW),
// matching the [128][40] bf16 LDS layout (row stride 80B).
// This toolchain's builtin is the 6-arg form:
//   (uint32x4 g0, int32x8 g1, int32x4 g2, int32x4 g3, int32x8 g4, i32 cpol)
static __device__ inline void tdm_load_2d(uint32_t lds_addr, uint64_t gaddr,
                                          uint32_t tensor_d0, uint32_t tensor_d1,
                                          uint64_t stride0,
                                          uint32_t tile_d0, uint32_t tile_d1) {
    union { v4u v; uint64_t u[2]; } g0;
    g0.u[0] = 1ull /*count=1*/ | ((uint64_t)lds_addr << 32);
    g0.u[1] = (gaddr & ((1ull << 57) - 1)) | (2ull << 62);   // type=2 "image"

    union { v8i v; uint64_t u[4]; } g1;
    uint64_t w0 = ((uint64_t)2 << 16)            // data_size = 4B units
                | (1ull << 20)                   // pad_enable
                | (3ull << 22)                   // pad_interval: 16 DWORDs
                | (3ull << 25);                  // pad_amount : 4 DWORDs
    g1.u[0] = w0 | ((uint64_t)(tensor_d0 & 0xFFFFu) << 48);
    g1.u[1] = (uint64_t)(tensor_d0 >> 16)
            | ((uint64_t)tensor_d1 << 16)
            | ((uint64_t)(tile_d0 & 0xFFFFu) << 48);
    g1.u[2] = (uint64_t)(tile_d1 & 0xFFFFu)      // tile_dim1 (tile_dim2 = 0)
            | ((stride0 & 0xFFFFFFFFull) << 32); // tensor_dim0_stride[31:0]
    g1.u[3] = ((stride0 >> 32) & 0xFFFFull);     // stride[47:32]; dim1_stride=0

    v4i z4 = {0, 0, 0, 0};                       // groups 2/3 unused (2D tile)
    v8i z8 = {0, 0, 0, 0, 0, 0, 0, 0};
    __builtin_amdgcn_tensor_load_to_lds(g0.v, g1.v, z4, z4, z8, 0);
}
#endif

// ---------------- conversion / weight-prep kernels -------------------------

__global__ void k_cvt_bf16(const float* __restrict__ in, __bf16* __restrict__ out, int n) {
    int i = blockIdx.x * blockDim.x + threadIdx.x;
    if (i < n) out[i] = (__bf16)in[i];
}

// Wt[n][k] = bf16(W[k][n])   (W is [K][N] row-major)
__global__ void k_transpose_w(const float* __restrict__ W, __bf16* __restrict__ Wt,
                              int K, int N) {
    int i = blockIdx.x * blockDim.x + threadIdx.x;
    if (i >= N * K) return;
    int n = i / K;
    int k = i - n * K;
    Wt[i] = (__bf16)W[(size_t)k * N + n];
}

// Fold the summed group axis into the weights:
// Wkt[n][k] = sum_g W1[k][g*128+n],  Wvt[n][k] = sum_g W1[k][512+g*128+n]
__global__ void k_w1_reduce(const float* __restrict__ W1,
                            __bf16* __restrict__ Wkt, __bf16* __restrict__ Wvt) {
    int i = blockIdx.x * blockDim.x + threadIdx.x;
    if (i >= HD_ * E_) return;
    int n = i / E_;          // 0..127  (head-dim column)
    int k = i - n * E_;      // 0..2047 (input dim)
    const int LD = 2 * G_ * HD_;  // 1024
    float sk = 0.f, sv = 0.f;
    #pragma unroll
    for (int g = 0; g < G_; ++g) {
        sk += W1[(size_t)k * LD + g * HD_ + n];
        sv += W1[(size_t)k * LD + G_ * HD_ + g * HD_ + n];
    }
    Wkt[(size_t)n * E_ + k] = (__bf16)sk;
    Wvt[(size_t)n * E_ + k] = (__bf16)sv;
}

// ---------------- tiled bf16 WMMA GEMM  C[M,N] = A[M,K] @ Bt[N,K]^T --------
// 256 threads = 8 waves; block tile 128x128; wave tile 64x32 (4x2 WMMA tiles)
// LDS staging via Tensor Data Mover (wave 0 issues, TENSORcnt sync).
// TRANSV: scatter output as C[(m/Sdim)*N + n][m%Sdim]  (per-batch transpose,
//         used to produce V^T so the PV matmul's B-fragments are contiguous)

template <typename OutT, bool TRANSV>
__global__ void __launch_bounds__(256)
k_gemm_bf16(const __bf16* __restrict__ A, const __bf16* __restrict__ Bt,
            OutT* __restrict__ C, int M, int N, int K, int Sdim) {
    __shared__ __bf16 lA[128][40];   // +8 bf16 pad: row stride 80B, conflict-free b128
    __shared__ __bf16 lB[128][40];

    const int tid   = threadIdx.x;
    const int lane  = tid & 31;
    const int wave  = tid >> 5;       // 0..7
    const int waveM = wave >> 2;      // 0..1 -> 64-row slab
    const int waveN = wave & 3;       // 0..3 -> 32-col slab
    const int mBase = blockIdx.y * 128;
    const int nBase = blockIdx.x * 128;
    const int hi    = lane >> 4;      // 0/1 : which 16-lane half
    const int l16   = lane & 15;
#if !HAVE_TDM
    const int row   = tid >> 1;       // staging: 2 threads/row, 16 bf16 each
    const int half  = tid & 1;
#endif

    v8f acc[4][2];
    #pragma unroll
    for (int mt = 0; mt < 4; ++mt)
        #pragma unroll
        for (int nt = 0; nt < 2; ++nt) acc[mt][nt] = vzero8();

    for (int k0 = 0; k0 < K; k0 += 32) {
        __syncthreads();
#if HAVE_TDM
        // ---- TDM: DMA the 128x32 bf16 A and B tiles into padded LDS ----
        if (wave == 0) {
            tdm_load_2d((uint32_t)(uintptr_t)&lA[0][0],
                        (uint64_t)(uintptr_t)A + ((uint64_t)mBase * K + k0) * 2,
                        (uint32_t)(K >> 1), 128u, (uint64_t)(K >> 1),
                        16u /*64B row*/, 128u);
            tdm_load_2d((uint32_t)(uintptr_t)&lB[0][0],
                        (uint64_t)(uintptr_t)Bt + ((uint64_t)nBase * K + k0) * 2,
                        (uint32_t)(K >> 1), 128u, (uint64_t)(K >> 1),
                        16u, 128u);
            __builtin_amdgcn_s_wait_tensorcnt(0);
        }
#else
        // ---- fallback: manual coalesced b128 staging ----
        {
            const uint4* sa = (const uint4*)&A[(size_t)(mBase + row) * K + k0 + half * 16];
            uint4* da = (uint4*)&lA[row][half * 16];
            da[0] = sa[0]; da[1] = sa[1];
            const uint4* sb = (const uint4*)&Bt[(size_t)(nBase + row) * K + k0 + half * 16];
            uint4* db = (uint4*)&lB[row][half * 16];
            db[0] = sb[0]; db[1] = sb[1];
        }
        if (k0 + 32 < K) {
            __builtin_prefetch(&A[(size_t)(mBase + row) * K + k0 + 32 + half * 16], 0, 1);
            __builtin_prefetch(&Bt[(size_t)(nBase + row) * K + k0 + 32 + half * 16], 0, 1);
        }
#endif
        __syncthreads();

        // ---- fragments from LDS, contiguous k = 16*hi + s for A and B ----
        FragBF aF[4], bF[2];
        #pragma unroll
        for (int mt = 0; mt < 4; ++mt) {
            const int r  = waveM * 64 + mt * 16 + l16;  // A row = lane&15
            const int ko = hi * 16;                     // 16 contiguous K values
            aF[mt].q[0] = *(const uint4*)&lA[r][ko];
            aF[mt].q[1] = *(const uint4*)&lA[r][ko + 8];
        }
        #pragma unroll
        for (int nt = 0; nt < 2; ++nt) {
            const int n  = waveN * 32 + nt * 16 + l16;  // B col = lane&15
            const int kb = hi * 16;                     // 16 contiguous K values
            bF[nt].q[0] = *(const uint4*)&lB[n][kb];
            bF[nt].q[1] = *(const uint4*)&lB[n][kb + 8];
        }
        #pragma unroll
        for (int mt = 0; mt < 4; ++mt)
            #pragma unroll
            for (int nt = 0; nt < 2; ++nt)
                acc[mt][nt] = __builtin_amdgcn_wmma_f32_16x16x32_bf16(
                    false, aF[mt].v, false, bF[nt].v, (short)0, acc[mt][nt],
                    false, false);
    }

    // ---- store (C layout: VGPR r -> row r + hi*8, col lane&15) ----
    #pragma unroll
    for (int mt = 0; mt < 4; ++mt)
        #pragma unroll
        for (int nt = 0; nt < 2; ++nt)
            #pragma unroll
            for (int r = 0; r < 8; ++r) {
                const int m = mBase + waveM * 64 + mt * 16 + r + hi * 8;
                const int n = nBase + waveN * 32 + nt * 16 + l16;
                const float v = acc[mt][nt][r];
                if constexpr (TRANSV)
                    C[((size_t)(m / Sdim) * N + n) * Sdim + (m % Sdim)] = (OutT)v;
                else
                    C[(size_t)m * N + n] = (OutT)v;
            }
}

// ---------------- flash attention, shared K/V across heads -----------------
// 128 threads = 4 waves; each wave owns 16 query rows; s-blocks of 32.

__global__ void __launch_bounds__(128)
k_attn(const __bf16* __restrict__ Q, const __bf16* __restrict__ Kb,
       const __bf16* __restrict__ Vt, const float* __restrict__ mask,
       __bf16* __restrict__ O) {
    __shared__ __bf16 ldsP[4][16][32];   // per-wave P tile (C-layout -> A-frag)

    const int lane  = threadIdx.x & 31;
    const int wave  = threadIdx.x >> 5;
    const int hi    = lane >> 4;
    const int l16   = lane & 15;
    const int b     = blockIdx.z;
    const int h     = blockIdx.y;
    const int tBase = blockIdx.x * 64 + wave * 16;

    // Preload the wave's Q tile (16x128) as 4 A-fragments, reused for all s.
    FragBF qF[4];
    {
        const int t = tBase + l16;
        const size_t base = ((size_t)(b * S_) + t) * E_ + h * HD_;
        #pragma unroll
        for (int es = 0; es < 4; ++es) {
            const int ko = es * 32 + hi * 16;
            qF[es].q[0] = *(const uint4*)&Q[base + ko];
            qF[es].q[1] = *(const uint4*)&Q[base + ko + 8];
        }
    }

    v8f acc[8];                       // O tile 16x128 (8 col tiles)
    #pragma unroll
    for (int c = 0; c < 8; ++c) acc[c] = vzero8();
    float mrow[8], lrow[8];
    #pragma unroll
    for (int r = 0; r < 8; ++r) { mrow[r] = -INFINITY; lrow[r] = 0.f; }

    for (int s0 = 0; s0 < S_; s0 += 32) {
        // ---- S tile: Q(16x128) @ K^T -> two 16x16 score tiles ----
        v8f sAcc[2] = {vzero8(), vzero8()};
        #pragma unroll
        for (int j = 0; j < 2; ++j) {
            const int n = s0 + j * 16 + l16;                 // key row
            const size_t kr = ((size_t)(b * S_) + n) * HD_;  // contiguous in e
            #pragma unroll
            for (int es = 0; es < 4; ++es) {
                FragBF kF;
                const int kb = es * 32 + hi * 16;
                kF.q[0] = *(const uint4*)&Kb[kr + kb];
                kF.q[1] = *(const uint4*)&Kb[kr + kb + 8];
                sAcc[j] = __builtin_amdgcn_wmma_f32_16x16x32_bf16(
                    false, qF[es].v, false, kF.v, (short)0, sAcc[j], false, false);
            }
        }
        // ---- + mask ----
        #pragma unroll
        for (int r = 0; r < 8; ++r) {
            const int t = tBase + r + hi * 8;
            const size_t mb = ((size_t)b * S_ + t) * S_ + s0;
            sAcc[0][r] += mask[mb + l16];
            sAcc[1][r] += mask[mb + 16 + l16];
        }
        // ---- online softmax (row stats via intra-half shuffles) ----
        float pscale[8];
        #pragma unroll
        for (int r = 0; r < 8; ++r) {
            float v = fmaxf(sAcc[0][r], sAcc[1][r]);
            v = fmaxf(v, __shfl_xor(v, 1, 32));
            v = fmaxf(v, __shfl_xor(v, 2, 32));
            v = fmaxf(v, __shfl_xor(v, 4, 32));
            v = fmaxf(v, __shfl_xor(v, 8, 32));
            const float newm = fmaxf(mrow[r], v);
            const float sc = __expf(mrow[r] - newm);
            const float p0 = __expf(sAcc[0][r] - newm);
            const float p1 = __expf(sAcc[1][r] - newm);
            float rs = p0 + p1;
            rs += __shfl_xor(rs, 1, 32);
            rs += __shfl_xor(rs, 2, 32);
            rs += __shfl_xor(rs, 4, 32);
            rs += __shfl_xor(rs, 8, 32);
            lrow[r] = lrow[r] * sc + rs;
            mrow[r] = newm;
            pscale[r] = sc;
            sAcc[0][r] = p0;
            sAcc[1][r] = p1;
        }
        #pragma unroll
        for (int c = 0; c < 8; ++c)
            #pragma unroll
            for (int r = 0; r < 8; ++r) acc[c][r] *= pscale[r];

        // ---- C-layout P -> A-fragment via per-wave LDS tile ----
        #pragma unroll
        for (int r = 0; r < 8; ++r) {
            ldsP[wave][r + hi * 8][l16]      = (__bf16)sAcc[0][r];
            ldsP[wave][r + hi * 8][16 + l16] = (__bf16)sAcc[1][r];
        }
        __syncthreads();
        FragBF pF;
        {
            const int ko = hi * 16;
            pF.q[0] = *(const uint4*)&ldsP[wave][l16][ko];
            pF.q[1] = *(const uint4*)&ldsP[wave][l16][ko + 8];
        }
        // ---- O += P(16x32) @ V(32x128), V stored transposed [e][s] ----
        #pragma unroll
        for (int c = 0; c < 8; ++c) {
            FragBF vF;
            const int e = c * 16 + l16;
            const size_t vb = ((size_t)(b * HD_) + e) * S_ + s0 + hi * 16;
            vF.q[0] = *(const uint4*)&Vt[vb];
            vF.q[1] = *(const uint4*)&Vt[vb + 8];
            acc[c] = __builtin_amdgcn_wmma_f32_16x16x32_bf16(
                false, pF.v, false, vF.v, (short)0, acc[c], false, false);
        }
        __syncthreads();
    }

    // ---- normalize by row sums and store O (bf16, [b*S+t][h*128+e]) ----
    #pragma unroll
    for (int c = 0; c < 8; ++c)
        #pragma unroll
        for (int r = 0; r < 8; ++r) {
            const int t = tBase + r + hi * 8;
            const int e = c * 16 + l16;
            const float v = acc[c][r] / lrow[r];
            O[((size_t)(b * S_) + t) * E_ + h * HD_ + e] = (__bf16)v;
        }
}

// ---------------------------------------------------------------------------

extern "C" void kernel_launch(void* const* d_in, const int* in_sizes, int n_in,
                              void* d_out, int out_size, void* d_ws, size_t ws_size,
                              hipStream_t stream) {
    const float* x    = (const float*)d_in[0];
    const float* mask = (const float*)d_in[1];
    const float* W1   = (const float*)d_in[2];
    const float* W2   = (const float*)d_in[3];
    const float* W3   = (const float*)d_in[4];
    float* out = (float*)d_out;

    char* ws = (char*)d_ws;
    __bf16* xh   = (__bf16*)ws;  ws += (size_t)MT_ * E_ * 2;        // 16.8 MB
    __bf16* W2t  = (__bf16*)ws;  ws += (size_t)E_ * E_ * 2;         //  8.4 MB
    __bf16* W3t  = (__bf16*)ws;  ws += (size_t)E_ * E_ * 2;         //  8.4 MB
    __bf16* W1kt = (__bf16*)ws;  ws += (size_t)HD_ * E_ * 2;        //  0.5 MB
    __bf16* W1vt = (__bf16*)ws;  ws += (size_t)HD_ * E_ * 2;        //  0.5 MB
    __bf16* Qb   = (__bf16*)ws;  ws += (size_t)MT_ * E_ * 2;        // 16.8 MB
    __bf16* Kbuf = (__bf16*)ws;  ws += (size_t)MT_ * HD_ * 2;       //  1.0 MB
    __bf16* Vtb  = (__bf16*)ws;  ws += (size_t)B_ * HD_ * S_ * 2;   //  1.0 MB
    __bf16* Ob   = (__bf16*)ws;  ws += (size_t)MT_ * E_ * 2;        // 16.8 MB

    // 1) precision conversion + weight prep
    {
        int n = MT_ * E_;
        k_cvt_bf16<<<(n + 255) / 256, 256, 0, stream>>>(x, xh, n);
        int nw = E_ * E_;
        k_transpose_w<<<(nw + 255) / 256, 256, 0, stream>>>(W2, W2t, E_, E_);
        k_transpose_w<<<(nw + 255) / 256, 256, 0, stream>>>(W3, W3t, E_, E_);
        int nr = HD_ * E_;
        k_w1_reduce<<<(nr + 255) / 256, 256, 0, stream>>>(W1, W1kt, W1vt);
    }

    dim3 blk(256);
    // 2) Q = x @ W2        [4096,2048]
    k_gemm_bf16<__bf16, false><<<dim3(E_ / 128, MT_ / 128), blk, 0, stream>>>(
        xh, W2t, Qb, MT_, E_, E_, S_);
    // 3) Ksum = x @ W1k    [4096,128]
    k_gemm_bf16<__bf16, false><<<dim3(1, MT_ / 128), blk, 0, stream>>>(
        xh, W1kt, Kbuf, MT_, HD_, E_, S_);
    // 4) Vsum^T = (x @ W1v)^T  stored [b][e][s]
    k_gemm_bf16<__bf16, true><<<dim3(1, MT_ / 128), blk, 0, stream>>>(
        xh, W1vt, Vtb, MT_, HD_, E_, S_);
    // 5) attention
    k_attn<<<dim3(S_ / 64, H_, B_), dim3(128), 0, stream>>>(Qb, Kbuf, Vtb, mask, Ob);
    // 6) out = O @ W3 (fp32)
    k_gemm_bf16<float, false><<<dim3(E_ / 128, MT_ / 128), blk, 0, stream>>>(
        Ob, W3t, out, MT_, E_, E_, S_);
}
